// DenoisingGNN_51376398794888
// MI455X (gfx1250) — matrix operand order repaired and checked by reading, not compiled, not run on hardware
//
#include <hip/hip_runtime.h>
#include <hip/hip_bf16.h>

// ---------------------------------------------------------------------------
// DenoisingGNN forward for MI455X (gfx1250, wave32).
// Bandwidth-bound (~20 GFLOP vs ~0.6 GB @ 23.3 TB/s): fuse gather -> WMMA
// tile-GEMM -> attention-weighted atomic scatter per 16-edge tile.
// f16 WMMA inputs (v_wmma_f32_16x16x32_f16), f32 accumulation.
// WMMA operands live in fragment-major order (one aligned 32B access / lane).
// Edge gathers are issued on the CDNA5 async global->LDS engine
// (global_load_async_to_lds_b128, ASYNCcnt), which performs the gather AND
// the fragment permutation at 16B granularity off the wave's critical path.
// ---------------------------------------------------------------------------

typedef _Float16 v16h __attribute__((ext_vector_type(16)));
typedef float    v8f  __attribute__((ext_vector_type(8)));

#define I3 0.57735026918962576f   // 1/sqrt(3)
#define I6 0.40824829046386302f   // 1/sqrt(6)

__device__ inline v8f vzero8() {
  v8f z;
#pragma unroll
  for (int i = 0; i < 8; ++i) z[i] = 0.0f;
  return z;
}

__device__ inline v8f wmma16(v16h a, v16h b, v8f c) {
  // D = A(16x32 f16) * B(32x16 f16) + C(16x16 f32)
  return __builtin_amdgcn_wmma_f32_16x16x32_f16(false, a, false, b, (short)0, c,
                                                false, false);
}

// Position of element (m, krel) inside a fragment-major 16x32 A tile
// (ISA 7.12.2 wave32 A layout: lanes 0-15 -> M=lane, K {0..7,16..23};
//  lanes 16-31 -> M=lane-16, K {8..15,24..31}; 16 halves per lane).
__device__ inline int afrag_idx(int m, int krel) {
  int lane = m + (((krel >> 3) & 1) << 4);
  int j    = (krel & 7) + ((krel & 16) ? 8 : 0);
  return lane * 16 + j;
}

// One 32-byte aligned v16h fragment fetch (lowered to 2x b128 loads).
__device__ inline v16h load_frag(const _Float16* p, int lane) {
  return *(const v16h*)(p + lane * 16);
}

// 16-byte async copy global -> LDS (ASYNCcnt). Per-lane LDS destination
// address lets the async engine do gather + permutation in one shot.
__device__ inline void async_b128(const void* gsrc, void* ldst) {
  unsigned lds           = (unsigned)(size_t)ldst;   // low 32b = LDS offset
  unsigned long long gaddr = (unsigned long long)(size_t)gsrc;
  asm volatile("global_load_async_to_lds_b128 %0, %1, off"
               :: "v"(lds), "v"(gaddr)
               : "memory");
}
__device__ inline void wait_async0() {
  asm volatile("s_wait_asynccnt 0x0" ::: "memory");
}

// Monotonic float->uint encoding for atomic max over possibly-negative floats.
__device__ inline unsigned encf(float x) {
  unsigned u = __float_as_uint(x);
  return (u & 0x80000000u) ? ~u : (u | 0x80000000u);
}
__device__ inline float decf(unsigned u) {
  if (u == 0u) return 0.0f;  // untouched segment -> reference maps -inf to 0
  return (u & 0x80000000u) ? __uint_as_float(u & 0x7fffffffu)
                           : __uint_as_float(~u);
}

// ------------------------------- utility kernels ---------------------------

__global__ void k_zero(float* p, int n) {
  int i = blockIdx.x * blockDim.x + threadIdx.x;
  if (i < n) p[i] = 0.0f;
}

__global__ void k_cvt_f16(_Float16* d, const float* s, int n) {
  int i = blockIdx.x * blockDim.x + threadIdx.x;
  if (i < n) d[i] = (_Float16)s[i];
}

// Pack a row-major f32 weight (K x N) into f16 WMMA-B fragments:
// fragment (kt, nt) occupies 512 consecutive halves; within it, lane L holds
// 16 halves: lanes 0-15 -> N = nt*16+L, K = kt*32 + {0..7,16..23};
//            lanes 16-31 -> N = nt*16+L-16, K = kt*32 + {8..15,24..31}.
__global__ void k_packB(_Float16* dst, const float* src, int K, int N) {
  int idx = blockIdx.x * blockDim.x + threadIdx.x;
  if (idx >= K * N) return;
  int k = idx / N, n = idx - k * N;
  int kt = k >> 5, kr = k & 31;
  int nt = n >> 4, nr = n & 15;
  int lane = nr + (((kr >> 3) & 1) << 4);
  int j    = (kr & 7) + ((kr & 16) ? 8 : 0);
  dst[(size_t)((kt * (N >> 4) + nt) * 32 + lane) * 16 + j] = (_Float16)src[idx];
}

// ------------------------------- embedding ---------------------------------

__global__ void k_pool(const float* obj_at, const int* bmask, const int* obatch,
                       const float* W_emb, float* pooled, float* cnt, int Nn) {
  int idx = blockIdx.x * blockDim.x + threadIdx.x;
  if (idx >= Nn * 128) return;
  int n = idx >> 7, k = idx & 127;
  if (bmask[n] != 0) return;  // backbone -> not conditional
  float v = 0.0f;
#pragma unroll
  for (int t = 0; t < 4; ++t) v += obj_at[n * 4 + t] * W_emb[t * 128 + k];
  v *= 0.5f;  // 1/sqrt(NT)
  int g = obatch[n];
  atomicAdd(&pooled[g * 128 + k], v);
  if (k == 0) atomicAdd(&cnt[g], 1.0f);
}

__global__ void k_graph(const int* timestep, const float* pooled,
                        const float* cnt, const float* Wt, const float* bt,
                        const float* Wc, const float* bc, float* gproj, int G) {
  int k = threadIdx.x;  // 128 threads
  for (int g = 0; g < G; ++g) {
    float t  = (float)timestep[g];
    float cn = fmaxf(cnt[g], 1.0f);
    float acc = bt[k] + bc[k];
    for (int j = 0; j < 128; ++j) {
      int jj   = (j < 64) ? j : (j - 64);
      float fr = __expf(-9.210340371976184f * (float)jj / 64.0f);
      float te = (j < 64) ? __cosf(t * fr) : __sinf(t * fr);
      acc += te * Wt[j * 128 + k] + (pooled[g * 128 + j] / cn) * Wc[j * 128 + k];
    }
    gproj[g * 128 + k] = acc;
  }
}

__global__ void k_embed(const float* at, const int* bmap, const float* W_emb,
                        const float* gproj, float* s, int Nn) {
  int idx = blockIdx.x * blockDim.x + threadIdx.x;
  if (idx >= Nn * 128) return;
  int n = idx >> 7, k = idx & 127;
  float v = 0.0f;
#pragma unroll
  for (int t = 0; t < 4; ++t) v += at[n * 4 + t] * W_emb[t * 128 + k];
  s[idx] = 0.5f * v + gproj[bmap[n] * 128 + k];
}

__global__ void k_sh1(const float* pos, const int* row, const int* col,
                      float* sh1, int E) {
  int e = blockIdx.x * blockDim.x + threadIdx.x;
  if (e >= E) return;
  int r = row[e], c = col[e];
  float x = pos[r * 3 + 0] - pos[c * 3 + 0];
  float y = pos[r * 3 + 1] - pos[c * 3 + 1];
  float z = pos[r * 3 + 2] - pos[c * 3 + 2];
  float nrm = sqrtf(x * x + y * y + z * z) + 1e-12f;
  float s3  = 1.7320508075688772f / nrm;
  sh1[e * 3 + 0] = s3 * x;
  sh1[e * 3 + 1] = s3 * y;
  sh1[e * 3 + 2] = s3 * z;
}

// ------------------------------- attention ---------------------------------

__global__ void k_logit(const _Float16* s16, const int* row, const int* col,
                        const float* Wa, const float* ba, float* logit,
                        unsigned* menc, int S, int E) {
  int e    = blockIdx.x * 8 + (threadIdx.x >> 5);
  int lane = threadIdx.x & 31;
  if (e >= E) return;
  int r = row[e], c = col[e];
  float acc = 0.0f;
  for (int k = lane; k < S; k += 32)
    acc += (float)s16[(size_t)r * S + k] * Wa[k] +
           (float)s16[(size_t)c * S + k] * Wa[S + k];
#pragma unroll
  for (int off = 16; off; off >>= 1) acc += __shfl_down(acc, off, 32);
  if (lane == 0) {
    float x = acc + ba[0];
    x = (x >= 0.0f) ? x : 0.2f * x;
    logit[e] = x;
    atomicMax(&menc[r], encf(x));
  }
}

__global__ void k_expsum(const float* logit_in, const unsigned* menc,
                         const int* row, float* escal, float* den, int E) {
  int e = blockIdx.x * blockDim.x + threadIdx.x;
  if (e >= E) return;
  int r = row[e];
  float ex = __expf(logit_in[e] - decf(menc[r]));
  escal[e] = ex;
  atomicAdd(&den[r], ex);
}

__global__ void k_attw(float* escal, const float* den, const int* row, int E) {
  int e = blockIdx.x * blockDim.x + threadIdx.x;
  if (e >= E) return;
  escal[e] = escal[e] / (den[row[e]] + 1e-16f);
}

// ------------------------- edge messages (WMMA core) -----------------------
// One block = 16 edges, 256 threads = 8 waves.
// ms = a0*(xs@w1 + i3*(xv.sh1)@w2)                          -> scatter ags
// mv = a1*(i3*(xs@w3)*sh1 + i3*xv@w4 + i6*cross(xv,sh1)@w5) -> scatter agv
__global__ void k_msg(const _Float16* s16, const float* vv, const float* sh1,
                      const int* row, const int* col, const _Float16* w1p,
                      const _Float16* w2p, const _Float16* w3p,
                      const _Float16* w4p, const _Float16* w5p,
                      const float* att, float* ags, float* agv, int S,
                      float a0, float a1) {
  __shared__ __align__(32) _Float16 xsfrag[4 * 512];  // up to 4 K-fragments
  __shared__ __align__(32) _Float16 t2frag[512];
  __shared__ __align__(32) _Float16 xvfrag[3][512];
  __shared__ __align__(32) _Float16 crfrag[3][512];
  __shared__ __align__(16) float xvf[16][96];
  __shared__ float t3f[16][32];
  __shared__ int   cls[16], rws[16];
  __shared__ float atw[16], shv[16][3];

  int tid = threadIdx.x;
  int e0  = blockIdx.x * 16;
  // Warm the packed weight panels into WGP$/L2 (global_prefetch_b8).
  __builtin_prefetch(w1p + (size_t)tid * 64, 0, 0);
  if (tid < 16) {
    int e = e0 + tid;
    cls[tid] = col[e];
    rws[tid] = row[e];
    atw[tid] = att[e];
    shv[tid][0] = sh1[e * 3 + 0];
    shv[tid][1] = sh1[e * 3 + 1];
    shv[tid][2] = sh1[e * 3 + 2];
  }
  __syncthreads();
  // Async gather: scalar features (f16) straight into WMMA-A fragment order.
  // A lane-octet (8 consecutive K halves) is contiguous in both layouts.
  int spe = S >> 3;  // 16B copies per edge
  for (int idx = tid; idx < 16 * spe; idx += 256) {
    int e = idx / spe, oct = idx - e * spe;
    int k0 = oct << 3;
    int kt = k0 >> 5, krel = k0 & 31;
    int lane16 = e + (((krel >> 3) & 1) << 4);
    int j      = (krel & 16) ? 8 : 0;
    async_b128(s16 + (size_t)cls[e] * S + k0,
               xsfrag + kt * 512 + lane16 * 16 + j);
  }
  // Async gather: vector features (f32 rows, 24 x 16B per edge).
  for (int idx = tid; idx < 16 * 24; idx += 256) {
    int e = idx / 24, q = idx - e * 24;
    async_b128(vv + (size_t)cls[e] * 96 + q * 4, &xvf[e][q * 4]);
  }
  wait_async0();
  __syncthreads();
  for (int idx = tid; idx < 16 * 32; idx += 256) {
    int e = idx >> 5, u = idx & 31;
    float x0 = xvf[e][u * 3 + 0], x1 = xvf[e][u * 3 + 1], x2 = xvf[e][u * 3 + 2];
    float s0 = shv[e][0], s1 = shv[e][1], s2 = shv[e][2];
    int fi = afrag_idx(e, u);
    t2frag[fi]    = (_Float16)(I3 * (x0 * s0 + x1 * s1 + x2 * s2));
    xvfrag[0][fi] = (_Float16)(I3 * x0);
    xvfrag[1][fi] = (_Float16)(I3 * x1);
    xvfrag[2][fi] = (_Float16)(I3 * x2);
    crfrag[0][fi] = (_Float16)(I6 * (x1 * s2 - x2 * s1));
    crfrag[1][fi] = (_Float16)(I6 * (x2 * s0 - x0 * s2));
    crfrag[2][fi] = (_Float16)(I6 * (x0 * s1 - x1 * s0));
  }
  __syncthreads();

  int wave = tid >> 5, lane = tid & 31;
  int nms = S >> 4;  // 16-wide output tiles of ms
  // Phase A: ms tiles + t3 = xs@w3 (needed by phase B)
  for (int job = wave; job < nms + 2; job += 8) {
    v8f acc = vzero8();
    if (job < nms) {
      int n0 = job << 4;
      for (int k0 = 0; k0 < S; k0 += 32)
        acc = wmma16(load_frag(xsfrag + (k0 >> 5) * 512, lane),
                     load_frag(w1p + (size_t)((k0 >> 5) * nms + job) * 512, lane),
                     acc);
      acc = wmma16(load_frag(t2frag, lane),
                   load_frag(w2p + (size_t)job * 512, lane), acc);
      int n = n0 + (lane & 15), eb = (lane & 16) ? 8 : 0;
#pragma unroll
      for (int r = 0; r < 8; ++r) {
        int e = eb + r;
        atomicAdd(&ags[rws[e] * S + n], a0 * atw[e] * acc[r]);
      }
    } else {
      int nt = job - nms;  // 0,1 : t3 = xs @ w3  (N = 32)
      for (int k0 = 0; k0 < S; k0 += 32)
        acc = wmma16(load_frag(xsfrag + (k0 >> 5) * 512, lane),
                     load_frag(w3p + (size_t)((k0 >> 5) * 2 + nt) * 512, lane),
                     acc);
      int n = (nt << 4) + (lane & 15), eb = (lane & 16) ? 8 : 0;
#pragma unroll
      for (int r = 0; r < 8; ++r) t3f[eb + r][n] = acc[r];
    }
  }
  __syncthreads();
  // Phase B: mv[:, :, i] for i=0..2, two 16-wide N tiles each
  for (int job = wave; job < 6; job += 8) {
    int i = job >> 1, nt = job & 1;
    v8f acc = vzero8();
    acc = wmma16(load_frag(xvfrag[i], lane),
                 load_frag(w4p + (size_t)nt * 512, lane), acc);
    acc = wmma16(load_frag(crfrag[i], lane),
                 load_frag(w5p + (size_t)nt * 512, lane), acc);
    int n = (nt << 4) + (lane & 15), eb = (lane & 16) ? 8 : 0;
#pragma unroll
    for (int r = 0; r < 8; ++r) {
      int e = eb + r;
      float val = a1 * atw[e] * (acc[r] + I3 * t3f[e][n] * shv[e][i]);
      atomicAdd(&agv[rws[e] * 96 + n * 3 + i], val);
    }
  }
}

// ------------------- node linear + residual + BN column sums ---------------
__global__ void k_lin(const float* s_in, const float* v_in, const float* ags,
                      const float* agv, const _Float16* Wfsp,
                      const _Float16* Wfvp, float* s2, float* v2, float* musum,
                      float* vnsum, int S, float rsS, float rs32, int Nn) {
  __shared__ __align__(32) _Float16 asfrag[4 * 512];
  __shared__ __align__(32) _Float16 avfrag[3][512];
  int tid = threadIdx.x;
  int n0  = blockIdx.x * 16;
  __builtin_prefetch(Wfsp + (size_t)tid * 64, 0, 0);
  for (int idx = tid; idx < 16 * S; idx += 256) {
    int e = idx / S, k = idx - e * S;
    asfrag[(k >> 5) * 512 + afrag_idx(e, k & 31)] =
        (_Float16)ags[(n0 + e) * S + k];
  }
  for (int idx = tid; idx < 16 * 96; idx += 256) {
    int e = idx / 96, r = idx - e * 96;
    int u = r / 3, i = r - u * 3;
    avfrag[i][afrag_idx(e, u)] = (_Float16)agv[(n0 + e) * 96 + r];
  }
  __syncthreads();
  int wave = tid >> 5, lane = tid & 31;
  int nS = S >> 4;
  for (int job = wave; job < nS + 6; job += 8) {
    v8f acc = vzero8();
    if (job < nS) {
      int c0 = job << 4;
      for (int k0 = 0; k0 < S; k0 += 32)
        acc = wmma16(load_frag(asfrag + (k0 >> 5) * 512, lane),
                     load_frag(Wfsp + (size_t)((k0 >> 5) * nS + job) * 512, lane),
                     acc);
      int n = c0 + (lane & 15), eb = (lane & 16) ? 8 : 0;
      float part = 0.0f;
#pragma unroll
      for (int r = 0; r < 8; ++r) {
        int node  = n0 + eb + r;
        float val = acc[r] * rsS + s_in[node * S + n];
        s2[node * S + n] = val;
        part += val;
      }
      atomicAdd(&musum[n], part);
    } else {
      int j2 = job - nS;
      int i = j2 >> 1, nt = j2 & 1;
      acc = wmma16(load_frag(avfrag[i], lane),
                   load_frag(Wfvp + (size_t)nt * 512, lane), acc);
      int w = (nt << 4) + (lane & 15), eb = (lane & 16) ? 8 : 0;
      float part = 0.0f;
#pragma unroll
      for (int r = 0; r < 8; ++r) {
        int node  = n0 + eb + r;
        float val = acc[r] * rs32 + v_in[node * 96 + w * 3 + i];
        v2[node * 96 + w * 3 + i] = val;
        part += val * val;
      }
      atomicAdd(&vnsum[w], part);
    }
  }
}

__global__ void k_bnvar(const float* s2, const float* musum, float* varsum,
                        int S, int Nn) {
  int k   = threadIdx.x;  // blockDim.x == S
  float mu = musum[k] / (float)Nn;
  float loc = 0.0f;
  for (int n = blockIdx.x; n < Nn; n += gridDim.x) {
    float d = s2[n * S + k] - mu;
    loc += d * d;
  }
  atomicAdd(&varsum[k], loc);
}

__global__ void k_gate(const float* s2, const float* v2, const float* musum,
                       const float* varsum, const float* vnsum,
                       const float* bn_ws, const float* bn_bs,
                       const float* bn_wv, float* s_out, float* v_out, int S,
                       int n_plain, int Nn) {
  int idx = blockIdx.x * blockDim.x + threadIdx.x;
  if (idx >= Nn * S) return;
  int n = idx / S, k = idx - n * S;
  float mu  = musum[k] / (float)Nn;
  float var = varsum[k] / (float)Nn;
  float sc  = (s2[n * S + k] - mu) * bn_ws[k] * rsqrtf(var + 1e-5f) + bn_bs[k];
  if (k < n_plain) {
    s_out[n * n_plain + k] = fmaxf(sc, 0.0f);
  } else {
    int u    = k - n_plain;
    float g  = 1.0f / (1.0f + __expf(-sc));
    float vs = bn_wv[u] * rsqrtf(vnsum[u] / (3.0f * (float)Nn) + 1e-5f);
#pragma unroll
    for (int i = 0; i < 3; ++i)
      v_out[n * 96 + u * 3 + i] = v2[n * 96 + u * 3 + i] * vs * g;
  }
}

// ------------------------------- output heads ------------------------------
__global__ void k_head(const float* s, const float* v, const float* Wo_s,
                       const float* Wo_v, float* out, float* flat, int Nn) {
  int n = blockIdx.x * blockDim.x + threadIdx.x;
  if (n >= Nn) return;
#pragma unroll
  for (int k = 0; k < 4; ++k) {
    float acc = 0.0f;
    for (int j = 0; j < 64; ++j) acc += s[n * 64 + j] * Wo_s[j * 4 + k];
    out[n * 7 + k] = acc * 0.125f;  // 1/sqrt(64)
  }
#pragma unroll
  for (int i = 0; i < 3; ++i) {
    float acc = 0.0f;
    for (int u = 0; u < 32; ++u) acc += v[n * 96 + u * 3 + i] * Wo_v[u];
    out[n * 7 + 4 + i] = acc * 0.17677669529663689f;  // 1/sqrt(32)
  }
  for (int j = 0; j < 64; ++j) flat[n * 160 + j] = s[n * 64 + j];
  for (int r = 0; r < 96; ++r) flat[n * 160 + 64 + r] = v[n * 96 + r];
}

__global__ void k_edge(const float* flat, const int* ec, const float* We,
                       const float* be, float* out, int EC) {
  int e    = blockIdx.x * 8 + (threadIdx.x >> 5);
  int lane = threadIdx.x & 31;
  if (e >= EC) return;
  int rc = ec[e], cc = ec[EC + e];
  float acc = 0.0f;
  for (int k = lane; k < 128; k += 32)
    acc += flat[rc * 160 + k] * We[k] + flat[cc * 160 + k] * We[128 + k];
#pragma unroll
  for (int off = 16; off; off >>= 1) acc += __shfl_down(acc, off, 32);
  if (lane == 0) out[e] = acc + be[0];
}

// ------------------------------- host driver -------------------------------
static inline int cdiv(int a, int b) { return (a + b - 1) / b; }

extern "C" void kernel_launch(void* const* d_in, const int* in_sizes, int n_in,
                              void* d_out, int out_size, void* d_ws,
                              size_t ws_size, hipStream_t stream) {
  (void)n_in; (void)out_size; (void)ws_size;
  const float* atom_types = (const float*)d_in[0];
  const float* pos        = (const float*)d_in[1];
  const int*   timestep   = (const int*)d_in[2];
  const float* obj_at     = (const float*)d_in[3];
  const int*   eidx       = (const int*)d_in[4];
  const int*   bmap       = (const int*)d_in[5];
  const int*   bmask      = (const int*)d_in[6];
  const int*   obatch     = (const int*)d_in[7];
  const int*   ecand      = (const int*)d_in[8];
  const float* W_emb      = (const float*)d_in[9];
  const float* Wt = (const float*)d_in[10]; const float* bt = (const float*)d_in[11];
  const float* Wc = (const float*)d_in[12]; const float* bc = (const float*)d_in[13];
  const float* cw[2][5]  = {{(const float*)d_in[14], (const float*)d_in[15],
                             (const float*)d_in[16], (const float*)d_in[17],
                             (const float*)d_in[18]},
                            {(const float*)d_in[26], (const float*)d_in[27],
                             (const float*)d_in[28], (const float*)d_in[29],
                             (const float*)d_in[30]}};
  const float* cWa[2]  = {(const float*)d_in[19], (const float*)d_in[31]};
  const float* cba[2]  = {(const float*)d_in[20], (const float*)d_in[32]};
  const float* cWfs[2] = {(const float*)d_in[21], (const float*)d_in[33]};
  const float* cWfv[2] = {(const float*)d_in[22], (const float*)d_in[34]};
  const float* cbnw[2] = {(const float*)d_in[23], (const float*)d_in[35]};
  const float* cbnb[2] = {(const float*)d_in[24], (const float*)d_in[36]};
  const float* cbnv[2] = {(const float*)d_in[25], (const float*)d_in[37]};
  const float* Wo_s = (const float*)d_in[38];
  const float* Wo_v = (const float*)d_in[39];
  const float* We   = (const float*)d_in[40];
  const float* be   = (const float*)d_in[41];

  const int Nn = in_sizes[0] / 4;
  const int E  = in_sizes[4] / 2;
  const int EC = in_sizes[8] / 2;
  const int G  = in_sizes[2];
  const int*   row = eidx;
  const int*   col = eidx + E;
  float*       out = (float*)d_out;

  // workspace layout (floats)
  float* W = (float*)d_ws;
  size_t o = 0;
  auto nxt = [&](size_t n) { float* p = W + o; o += n; return p; };
  float* s_cur  = nxt((size_t)Nn * 128);
  float* v_cur  = nxt((size_t)Nn * 96);
  float* s2     = nxt((size_t)Nn * 128);
  float* v2     = nxt((size_t)Nn * 96);
  float* ags    = nxt((size_t)Nn * 128);
  float* agv    = nxt((size_t)Nn * 96);
  float* sh1    = nxt((size_t)E * 3);
  float* escal  = nxt((size_t)E);
  float* mencf  = nxt((size_t)Nn);
  float* den    = nxt((size_t)Nn);
  float* pooled = nxt((size_t)G * 128);
  float* cnt    = nxt((size_t)G);
  float* gproj  = nxt((size_t)G * 128);
  float* musum  = nxt(128);
  float* varsum = nxt(128);
  float* vnsum  = nxt(32);
  float* flat   = nxt((size_t)Nn * 160);
  o = (o + 7) & ~(size_t)7;  // 32-byte align the packed f16 region
  _Float16* Hb   = (_Float16*)(W + o);
  _Float16* w1p  = Hb;           // 128*128
  _Float16* w2p  = w1p + 16384;  // 32*128
  _Float16* w3p  = w2p + 4096;   // 128*32
  _Float16* w4p  = w3p + 4096;   // 32*32
  _Float16* w5p  = w4p + 1024;   // 32*32
  _Float16* Wfsp = w5p + 1024;   // 128*128
  _Float16* Wfvp = Wfsp + 16384; // 32*32
  _Float16* s16  = Wfvp + 1024;  // Nn*128 f16 mirror of node scalars
  unsigned* menc = (unsigned*)mencf;

  const int TB = 256;
  // ---- embedding ----
  k_zero<<<cdiv(G * 128 + G, TB), TB, 0, stream>>>(pooled, G * 128 + G);
  k_pool<<<cdiv(Nn * 128, TB), TB, 0, stream>>>(obj_at, bmask, obatch, W_emb,
                                                pooled, cnt, Nn);
  k_graph<<<1, 128, 0, stream>>>(timestep, pooled, cnt, Wt, bt, Wc, bc, gproj, G);
  k_embed<<<cdiv(Nn * 128, TB), TB, 0, stream>>>(atom_types, bmap, W_emb, gproj,
                                                 s_cur, Nn);
  k_zero<<<cdiv(Nn * 96, TB), TB, 0, stream>>>(v_cur, Nn * 96);
  k_sh1<<<cdiv(E, TB), TB, 0, stream>>>(pos, row, col, sh1, E);

  // ---- two equivariant conv layers ----
  for (int L = 0; L < 2; ++L) {
    const int S       = (L == 0) ? 128 : 96;
    const int n_plain = (L == 0) ? 96 : 64;
    const float a0   = rsqrtf((float)(S + 32));
    const float a1   = sqrtf(3.0f / (float)(S + 64));
    const float rsS  = rsqrtf((float)S);
    const float rs32 = rsqrtf(32.0f);
    // pack weights into f16 WMMA-B fragment order; mirror node scalars to f16
    k_packB<<<cdiv(S * S, TB), TB, 0, stream>>>(w1p, cw[L][0], S, S);
    k_packB<<<cdiv(32 * S, TB), TB, 0, stream>>>(w2p, cw[L][1], 32, S);
    k_packB<<<cdiv(S * 32, TB), TB, 0, stream>>>(w3p, cw[L][2], S, 32);
    k_packB<<<cdiv(1024, TB), TB, 0, stream>>>(w4p, cw[L][3], 32, 32);
    k_packB<<<cdiv(1024, TB), TB, 0, stream>>>(w5p, cw[L][4], 32, 32);
    k_packB<<<cdiv(S * S, TB), TB, 0, stream>>>(Wfsp, cWfs[L], S, S);
    k_packB<<<cdiv(1024, TB), TB, 0, stream>>>(Wfvp, cWfv[L], 32, 32);
    k_cvt_f16<<<cdiv(Nn * S, TB), TB, 0, stream>>>(s16, s_cur, Nn * S);
    // clear accumulators
    k_zero<<<cdiv(2 * Nn, TB), TB, 0, stream>>>(mencf, 2 * Nn);  // menc + den
    k_zero<<<cdiv(Nn * S, TB), TB, 0, stream>>>(ags, Nn * S);
    k_zero<<<cdiv(Nn * 96, TB), TB, 0, stream>>>(agv, Nn * 96);
    k_zero<<<1, 288, 0, stream>>>(musum, 288);  // musum+varsum+vnsum contiguous
    // attention softmax over incoming edges of each destination
    k_logit<<<cdiv(E, 8), TB, 0, stream>>>(s16, row, col, cWa[L], cba[L],
                                           escal, menc, S, E);
    k_expsum<<<cdiv(E, TB), TB, 0, stream>>>(escal, menc, row, escal, den, E);
    k_attw<<<cdiv(E, TB), TB, 0, stream>>>(escal, den, row, E);
    // fused async-gather -> WMMA messages -> weighted scatter
    k_msg<<<E / 16, TB, 0, stream>>>(s16, v_cur, sh1, row, col, w1p, w2p, w3p,
                                     w4p, w5p, escal, ags, agv, S, a0, a1);
    // node linear + residual + BN stats
    k_lin<<<Nn / 16, TB, 0, stream>>>(s_cur, v_cur, ags, agv, Wfsp, Wfvp, s2,
                                      v2, musum, vnsum, S, rsS, rs32, Nn);
    k_bnvar<<<40, S, 0, stream>>>(s2, musum, varsum, S, Nn);
    k_gate<<<cdiv(Nn * S, TB), TB, 0, stream>>>(s2, v2, musum, varsum, vnsum,
                                                cbnw[L], cbnb[L], cbnv[L],
                                                s_cur, v_cur, S, n_plain, Nn);
  }

  // ---- output heads ----
  k_head<<<cdiv(Nn, TB), TB, 0, stream>>>(s_cur, v_cur, Wo_s, Wo_v, out, flat, Nn);
  k_edge<<<cdiv(EC, 8), TB, 0, stream>>>(flat, ecand, We, be, out + Nn * 7, EC);
}
